// MultiHeadAttentionQuantum_65481071407075
// MI455X (gfx1250) — matrix operand order repaired and checked by reading, not compiled, not run on hardware
//
#include <hip/hip_runtime.h>

typedef __attribute__((ext_vector_type(16))) _Float16 v16h;
typedef __attribute__((ext_vector_type(8)))  _Float16 v8h;
typedef __attribute__((ext_vector_type(8)))  float    v8f;

// Problem constants (from reference): B=64, S=8192, H=8, DK=8, E=64
#define NBATCH 64
#define SEQ    8192
#define GROUPS_PER_BATCH (SEQ / 16)      // 512 groups of 16 tokens
#define TOTAL_GROUPS (NBATCH * GROUPS_PER_BATCH)   // 32768
#define WAVES_PER_BLOCK 8
#define GROUPS_PER_WAVE 4
#define NBLOCKS (TOTAL_GROUPS / (WAVES_PER_BLOCK * GROUPS_PER_WAVE))  // 1024

__global__ __launch_bounds__(256)
void qattn_fused(const float* __restrict__ x,
                 const float* __restrict__ theta,
                 const float* __restrict__ w,      // [64,64] row-major: w[e][k]
                 const float* __restrict__ bias,   // [64]
                 float* __restrict__ y)            // [B, 8192, 64]
{
    __shared__ _Float16 wsh[64 * 64];                       // 8 KB, f16 weights
    __shared__ float    zbuf[WAVES_PER_BLOCK][16 * 8 * 8];  // 32 KB, z per wave
    __shared__ _Float16 atile[WAVES_PER_BLOCK][16 * 64];    // 16 KB, A tiles

    const int tid = threadIdx.x;
    const int wv  = tid >> 5;
    const int l   = tid & 31;
    const int lhi = l >> 4;     // 0: lanes 0-15, 1: lanes 16-31
    const int llo = l & 15;

    // Stage w_combine into LDS as f16 (row-major, wsh[e*64+k])
    for (int k = tid; k < 64 * 64; k += 256) wsh[k] = (_Float16)w[k];

    float th[8];
#pragma unroll
    for (int i = 0; i < 8; ++i) th[i] = theta[i];

    __syncthreads();

    // B fragments: B[k][n] = w[n][k].  contiguous k order used here (same as A):
    // lanes 0-15 hold K=0..15 (elem e -> K=e), lanes 16-31 hold K=16..31.
    v16h bfrag[4][2];
#pragma unroll
    for (int nt = 0; nt < 4; ++nt) {
        const int n = nt * 16 + llo;
#pragma unroll
        for (int kc = 0; kc < 2; ++kc) {
            const int kb = kc * 32 + lhi * 16;
            bfrag[nt][kc] = *(const v16h*)&wsh[n * 64 + kb];  // 32B contiguous
        }
    }

    float bvals[4];
#pragma unroll
    for (int nt = 0; nt < 4; ++nt) bvals[nt] = bias[nt * 16 + llo];

    const int wglobal = blockIdx.x * WAVES_PER_BLOCK + wv;

    for (int it = 0; it < GROUPS_PER_WAVE; ++it) {
        const int g  = wglobal * GROUPS_PER_WAVE + it;
        const int b  = g >> 9;           // / GROUPS_PER_BATCH
        const int gi = g & 511;
        const float* xg = x + (size_t)(b * SEQ + gi * 16) * 64;
        float*       yg = y + (size_t)b * SEQ * 64;

        // -------- phase 1: quantum heads z --------
#pragma unroll
        for (int rd = 0; rd < 4; ++rd) {
            const int lin = rd * 32 + l;
            const int t = lin >> 3, h = lin & 7;       // local token, head
            const float* xp = xg + (size_t)t * 64 + h * 8;
            if (it + 1 < GROUPS_PER_WAVE || g + 1 < TOTAL_GROUPS)
                __builtin_prefetch(xp + 16 * 64, 0, 0);  // next group's x
            const float4 xa = *(const float4*)xp;
            const float4 xb = *(const float4*)(xp + 4);
            float c[8];
            c[0] = __cosf(xa.x + th[0]); c[1] = __cosf(xa.y + th[1]);
            c[2] = __cosf(xa.z + th[2]); c[3] = __cosf(xa.w + th[3]);
            c[4] = __cosf(xb.x + th[4]); c[5] = __cosf(xb.y + th[5]);
            c[6] = __cosf(xb.z + th[6]); c[7] = __cosf(xb.w + th[7]);
            float oz[8];
            oz[0] = ((c[1] * c[2]) * (c[3] * c[4])) * ((c[5] * c[6]) * c[7]);
            float cp = c[0];
#pragma unroll
            for (int k = 1; k < 8; ++k) { cp *= c[k]; oz[k] = cp; }
            float* zp = &zbuf[wv][(t * 8 + h) * 8];
            *(float4*)zp       = make_float4(oz[0], oz[1], oz[2], oz[3]);
            *(float4*)(zp + 4) = make_float4(oz[4], oz[5], oz[6], oz[7]);
        }
        __syncthreads();

        // -------- phase 2: 8x8 attention across heads, per token --------
#pragma unroll
        for (int rd = 0; rd < 4; ++rd) {
            const int lin = rd * 32 + l;
            const int t = lin >> 3, hr = lin & 7;
            const float* zt = &zbuf[wv][t * 64];
            const float4 r0 = *(const float4*)(zt + hr * 8);
            const float4 r1 = *(const float4*)(zt + hr * 8 + 4);
            float sc[8];
#pragma unroll
            for (int tt = 0; tt < 8; ++tt) {
                const float4 q0 = *(const float4*)(zt + tt * 8);
                const float4 q1 = *(const float4*)(zt + tt * 8 + 4);
                sc[tt] = r0.x * q0.x + r0.y * q0.y + r0.z * q0.z + r0.w * q0.w
                       + r1.x * q1.x + r1.y * q1.y + r1.z * q1.z + r1.w * q1.w;
            }
            float m = sc[0];
#pragma unroll
            for (int tt = 1; tt < 8; ++tt) m = fmaxf(m, sc[tt]);
            float p[8], s = 0.f;
#pragma unroll
            for (int tt = 0; tt < 8; ++tt) {
                p[tt] = __expf((sc[tt] - m) * 0.35355339059327373f);  // 8^-0.5
                s += p[tt];
            }
            const float inv = 1.0f / s;
            float o[8] = {0.f, 0.f, 0.f, 0.f, 0.f, 0.f, 0.f, 0.f};
#pragma unroll
            for (int tt = 0; tt < 8; ++tt) {
                const float4 q0 = *(const float4*)(zt + tt * 8);
                const float4 q1 = *(const float4*)(zt + tt * 8 + 4);
                o[0] += p[tt] * q0.x; o[1] += p[tt] * q0.y;
                o[2] += p[tt] * q0.z; o[3] += p[tt] * q0.w;
                o[4] += p[tt] * q1.x; o[5] += p[tt] * q1.y;
                o[6] += p[tt] * q1.z; o[7] += p[tt] * q1.w;
            }
            // A-tile row M = h*2 + (t/8), cols (t%8)*8 + d  (reshape-compatible)
            v8h ov;
#pragma unroll
            for (int d = 0; d < 8; ++d) ov[d] = (_Float16)(o[d] * inv);
            *(v8h*)&atile[wv][(hr * 2 + (t >> 3)) * 64 + (t & 7) * 8] = ov;
        }
        __syncthreads();

        // -------- phase 3: combine GEMM via WMMA (16x64 @ 64x64 tile) --------
        // contiguous k order used here: lanes 0-15: elems {K0..15} of row M=lane;
        //                     lanes 16-31: elems {K16..31} of row M=lane-16.
        v16h afrag[2];
#pragma unroll
        for (int kc = 0; kc < 2; ++kc) {
            const int kb = kc * 32 + lhi * 16;
            const _Float16* ap = &atile[wv][llo * 64 + kb];
            const v8h alo = *(const v8h*)ap;
            const v8h ahi = *(const v8h*)(ap + 8);
            afrag[kc] = __builtin_shufflevector(alo, ahi,
                0, 1, 2, 3, 4, 5, 6, 7, 8, 9, 10, 11, 12, 13, 14, 15);
        }
#pragma unroll
        for (int nt = 0; nt < 4; ++nt) {
            v8f acc;
#pragma unroll
            for (int v = 0; v < 8; ++v) acc[v] = bvals[nt];  // bias (lane = N)
            acc = __builtin_amdgcn_wmma_f32_16x16x32_f16(
                false, afrag[0], false, bfrag[nt][0], (short)0, acc, false, false);
            acc = __builtin_amdgcn_wmma_f32_16x16x32_f16(
                false, afrag[1], false, bfrag[nt][1], (short)0, acc, false, false);
            // D layout: VGPR v -> row M = v + 8*lhi, lane -> col N = llo + nt*16
            // Output row r = (M>>1)*1024 + gi*2 + (M&1)
#pragma unroll
            for (int v = 0; v < 8; ++v) {
                const int M = v + lhi * 8;
                const int r = (M >> 1) * 1024 + gi * 2 + (M & 1);
                yg[(size_t)r * 64 + nt * 16 + llo] = acc[v];
            }
        }
        __syncthreads();
    }
}

extern "C" void kernel_launch(void* const* d_in, const int* in_sizes, int n_in,
                              void* d_out, int out_size, void* d_ws, size_t ws_size,
                              hipStream_t stream) {
    const float* x     = (const float*)d_in[0];
    const float* theta = (const float*)d_in[1];
    const float* w     = (const float*)d_in[2];
    const float* bias  = (const float*)d_in[3];
    float* y = (float*)d_out;
    hipLaunchKernelGGL(qattn_fused, dim3(NBLOCKS), dim3(256), 0, stream,
                       x, theta, w, bias, y);
}
